// Decoder_39101382262796
// MI455X (gfx1250) — matrix operand order repaired and checked by reading, not compiled, not run on hardware
//
#include <hip/hip_runtime.h>
#include <hip/hip_bf16.h>

// ---------------------------------------------------------------------------
// Problem constants (from reference)
// ---------------------------------------------------------------------------
#define BB   64
#define TT   31
#define SS   32
#define VV   34004
#define VVP  34048    // VV padded to multiple of 64 (logits N-strip)
#define EE   300
#define HH   512
#define MM   1024
#define CIN  812      // E + H
#define KZ   1324     // CIN + H  (combined K for gate GEMM)
#define KZP  1344     // KZ padded to multiple of 32
#define NZ   2048     // 4*H
#define RTOT 1984     // T*B rows of the final GEMM
#define RPAD 2048     // padded to multiple of 128

typedef __attribute__((ext_vector_type(16))) __bf16 v16bf;
typedef __attribute__((ext_vector_type(8)))  __bf16 v8bf;
typedef __attribute__((ext_vector_type(8)))  float  v8f;

// ---------------------------------------------------------------------------
// bf16 split helpers (bit manipulation only)
// ---------------------------------------------------------------------------
__device__ __forceinline__ __bf16 bits2bf(unsigned short s) {
    return __builtin_bit_cast(__bf16, s);
}
__device__ __forceinline__ __bf16 f2bf_rne(float x) {      // round-nearest-even
    unsigned u = __builtin_bit_cast(unsigned, x);
    unsigned r = u + 0x7FFFu + ((u >> 16) & 1u);
    return bits2bf((unsigned short)(r >> 16));
}
__device__ __forceinline__ float hi_part(float x) {        // truncated high half
    unsigned u = __builtin_bit_cast(unsigned, x) & 0xFFFF0000u;
    return __builtin_bit_cast(float, u);
}
__device__ __forceinline__ __bf16 bf_of_hi(float xh) {
    return bits2bf((unsigned short)(__builtin_bit_cast(unsigned, xh) >> 16));
}
__device__ __forceinline__ void split_store(float x, __bf16* hi, __bf16* lo, size_t o) {
    float xh = hi_part(x);
    hi[o] = bf_of_hi(xh);
    lo[o] = f2bf_rne(x - xh);
}

// ---------------------------------------------------------------------------
// Packed fragment loads (no conversion in GEMM inner loops)
//   A packed row-major [M][Kpad] bf16: lane m=lane&15; elements 0..15 are
//   k = kb..kb+15 contiguous (kb = (lane>>4)*16) -> two v8bf loads.
//   B packed transposed [N][Kpad] bf16: lane n=lane&15; elements 0..15 are
//   k = k0 + (lane>>4)*16 + 0..15 contiguous -> one v16bf (2x b128) load.
// ---------------------------------------------------------------------------
__device__ __forceinline__ v16bf frag_a(const __bf16* __restrict__ A, int lda,
                                        int m0, int k0, int lane) {
    const __bf16* p = A + (size_t)(m0 + (lane & 15)) * lda + k0 + ((lane >> 4) << 4);
    struct Pair { v8bf a, b; } q;
    q.a = *(const v8bf*)p;
    q.b = *(const v8bf*)(p + 8);
    return __builtin_bit_cast(v16bf, q);
}
__device__ __forceinline__ v16bf frag_b(const __bf16* __restrict__ Bt, int ldk,
                                        int n0, int k0, int lane) {
    const __bf16* p = Bt + (size_t)(n0 + (lane & 15)) * ldk + k0 + ((lane >> 4) << 4);
    return *(const v16bf*)p;
}

// 2-term split product: hi*hi + lo*hi + hi*lo (drops lo*lo ~ 2^-16 rel.)
__device__ __forceinline__ v8f wmma3(v8f acc, v16bf ah, v16bf al, v16bf bh, v16bf bl) {
    acc = __builtin_amdgcn_wmma_f32_16x16x32_bf16(false, ah, false, bh, (short)0, acc, false, false);
    acc = __builtin_amdgcn_wmma_f32_16x16x32_bf16(false, al, false, bh, (short)0, acc, false, false);
    acc = __builtin_amdgcn_wmma_f32_16x16x32_bf16(false, ah, false, bl, (short)0, acc, false, false);
    return acc;
}

__device__ __forceinline__ float sigm(float x) { return 1.0f / (1.0f + expf(-x)); }

// ---------------------------------------------------------------------------
// Pack kernels (one-time / per-step conversion, out of the GEMM hot loop)
// ---------------------------------------------------------------------------
// Row-major fp32 [n] -> bf16 hi/lo (same layout)
__global__ void pack_rows(const float* __restrict__ in, __bf16* __restrict__ hi,
                          __bf16* __restrict__ lo, int n) {
    int i = blockIdx.x * 256 + threadIdx.x;
    if (i < n) split_store(in[i], hi, lo, (size_t)i);
}

// Transpose-pack: virtual fp32 [K][N] (in1 rows < Ksplit, in2 rows Ksplit..K)
// -> bf16 hi/lo [Nalloc][Kpad], zero padded. Grid: (Nalloc/32, Kpad/32), 32x8.
// Source fp32 weights are dead after packing -> non-temporal loads so the
// 70 MB fp32 Wfc does not evict the L2-resident bf16 packs.
__global__ void pack_transpose(const float* __restrict__ in1, const float* __restrict__ in2,
                               int Ksplit, int K, int N, int Kpad,
                               __bf16* __restrict__ hi, __bf16* __restrict__ lo) {
    __shared__ float tile[32][33];
    int kb = blockIdx.y * 32, nb = blockIdx.x * 32;
#pragma unroll
    for (int i = 0; i < 4; ++i) {
        int k = kb + threadIdx.y + i * 8, n = nb + threadIdx.x;
        float x = 0.0f;
        if (n < N) {
            if (k < Ksplit)      x = __builtin_nontemporal_load(&in1[(size_t)k * N + n]);
            else if (k < K)      x = __builtin_nontemporal_load(&in2[(size_t)(k - Ksplit) * N + n]);
        }
        tile[threadIdx.y + i * 8][threadIdx.x] = x;
    }
    __syncthreads();
#pragma unroll
    for (int i = 0; i < 4; ++i) {
        int n = nb + threadIdx.y + i * 8, k = kb + threadIdx.x;
        split_store(tile[threadIdx.x][threadIdx.y + i * 8], hi, lo, (size_t)n * Kpad + k);
    }
}

// Per-step: u = [x_t | attn_prev | h] gathered + split into [64][KZP] bf16
__global__ void pack_u(const int* __restrict__ dec, const float* __restrict__ emb,
                       const float* __restrict__ attn_f32, const float* __restrict__ h_state,
                       __bf16* __restrict__ uhi, __bf16* __restrict__ ulo, int t) {
    int idx = blockIdx.x * 256 + threadIdx.x;            // 64 * 1344
    if (idx >= BB * KZP) return;
    int m = idx / KZP, k = idx - m * KZP;
    float x = 0.0f;
    if (k < EE)       x = emb[dec[m * TT + t] * EE + k];
    else if (k < CIN) x = t ? attn_f32[(t - 1) * (BB * HH) + m * HH + (k - EE)] : 0.0f;
    else if (k < KZ)  x = h_state[m * HH + (k - CIN)];
    split_store(x, uhi, ulo, (size_t)idx);
}

// ---------------------------------------------------------------------------
// GEMM 1: keys = memory @ Wm   [2048,1024] x [1024,512]
// 4 waves/block; wave w owns m-frag m0 = by*64 + 16w; 4 n-frags each.
// ---------------------------------------------------------------------------
__global__ __launch_bounds__(128)
void keys_gemm(const __bf16* __restrict__ Ah, const __bf16* __restrict__ Al,
               const __bf16* __restrict__ Bh, const __bf16* __restrict__ Bl,
               float* __restrict__ keys) {
    int lane = threadIdx.x & 31, w = threadIdx.x >> 5;
    int m0 = blockIdx.y * 64 + w * 16;
    int n0 = blockIdx.x * 64;
    v8f acc[4] = {};
    for (int k0 = 0; k0 < MM; k0 += 32) {
        v16bf ah = frag_a(Ah, MM, m0, k0, lane);
        v16bf al = frag_a(Al, MM, m0, k0, lane);
#pragma unroll
        for (int j = 0; j < 4; ++j) {
            v16bf bh = frag_b(Bh, MM, n0 + j * 16, k0, lane);
            v16bf bl = frag_b(Bl, MM, n0 + j * 16, k0, lane);
            acc[j] = wmma3(acc[j], ah, al, bh, bl);
        }
    }
    int n = lane & 15, mb = (lane >> 4) << 3;
#pragma unroll
    for (int j = 0; j < 4; ++j)
#pragma unroll
        for (int v = 0; v < 8; ++v)
            keys[(size_t)(m0 + mb + v) * HH + n0 + j * 16 + n] = acc[j][v];
}

// ---------------------------------------------------------------------------
// GEMM 2 (per step): z = u @ [Wk;Wr]   [64,1344] x [1344,2048]
// 4 waves/block cover all 64 rows; grid.x = 32 n-strips of 64.
// ---------------------------------------------------------------------------
__global__ __launch_bounds__(128)
void z_gemm(const __bf16* __restrict__ Ah, const __bf16* __restrict__ Al,
            const __bf16* __restrict__ Bh, const __bf16* __restrict__ Bl,
            float* __restrict__ z) {
    int lane = threadIdx.x & 31, w = threadIdx.x >> 5;
    int m0 = w * 16;
    int n0 = blockIdx.x * 64;
    v8f acc[4] = {};
    for (int k0 = 0; k0 < KZP; k0 += 32) {
        v16bf ah = frag_a(Ah, KZP, m0, k0, lane);
        v16bf al = frag_a(Al, KZP, m0, k0, lane);
#pragma unroll
        for (int j = 0; j < 4; ++j) {
            v16bf bh = frag_b(Bh, KZP, n0 + j * 16, k0, lane);
            v16bf bl = frag_b(Bl, KZP, n0 + j * 16, k0, lane);
            acc[j] = wmma3(acc[j], ah, al, bh, bl);
        }
    }
    int n = lane & 15, mb = (lane >> 4) << 3;
#pragma unroll
    for (int j = 0; j < 4; ++j)
#pragma unroll
        for (int v = 0; v < 8; ++v)
            z[(size_t)(m0 + mb + v) * NZ + n0 + j * 16 + n] = acc[j][v];
}

// ---------------------------------------------------------------------------
// Kernel 3 (per step): LSTM gate nonlinearity, update h/c in place
// ---------------------------------------------------------------------------
__global__ void lstm_kernel(const float* __restrict__ z, const float* __restrict__ bias,
                            float* __restrict__ h_state, float* __restrict__ c_state) {
    int idx = blockIdx.x * blockDim.x + threadIdx.x;     // 64*512
    int m = idx >> 9, j = idx & 511;
    const float* zr = z + (size_t)m * NZ;
    float zi = zr[j]          + bias[j];
    float zf = zr[j + HH]     + bias[j + HH];
    float zg = zr[j + 2 * HH] + bias[j + 2 * HH];
    float zo = zr[j + 3 * HH] + bias[j + 3 * HH];
    float c2 = sigm(zf) * c_state[idx] + sigm(zi) * tanhf(zg);
    float h2 = sigm(zo) * tanhf(c2);
    c_state[idx] = c2;
    h_state[idx] = h2;
}

// ---------------------------------------------------------------------------
// Kernel 4 (per step): Bahdanau attention + attention layer (fused per batch
// row). Also emits the bf16 hi/lo pack of attn2 for the final WMMA GEMM.
// ---------------------------------------------------------------------------
__global__ __launch_bounds__(512)
void attn_kernel(const float* __restrict__ h_state, const float* __restrict__ keys,
                 const float* __restrict__ Wq, const float* __restrict__ vvec,
                 const float* __restrict__ memory, const float* __restrict__ Wa,
                 float* __restrict__ attn_f32, __bf16* __restrict__ attn_hi,
                 __bf16* __restrict__ attn_lo, int t) {
    __shared__ float s_h2[HH];
    __shared__ float s_q[HH];
    __shared__ float s_part[512];
    __shared__ float s_al[SS];
    __shared__ float s_ctx[MM];

    int b = blockIdx.x, tid = threadIdx.x;
    s_h2[tid] = h_state[b * HH + tid];
    __syncthreads();

    float q = 0.0f;
    for (int h = 0; h < HH; ++h) q = fmaf(s_h2[h], Wq[h * HH + tid], q);
    s_q[tid] = q;
    __syncthreads();

    {   // score[s] = sum_h v[h]*tanh(keys+q); 16 threads per s
        int s = tid >> 4, l = tid & 15;
        const float* kr = keys + (size_t)b * (SS * HH) + s * HH;
        float p = 0.0f;
        for (int i = 0; i < 32; ++i) {
            int h = l + (i << 4);
            p = fmaf(vvec[h], tanhf(kr[h] + s_q[h]), p);
        }
        s_part[tid] = p;
    }
    __syncthreads();
    if (tid < SS) {
        float sum = 0.0f;
        for (int i = 0; i < 16; ++i) sum += s_part[tid * 16 + i];
        s_part[tid] = sum;
    }
    __syncthreads();
    if (tid == 0) {   // softmax over S=32
        float mx = -1e30f;
        for (int s = 0; s < SS; ++s) mx = fmaxf(mx, s_part[s]);
        float den = 0.0f;
        for (int s = 0; s < SS; ++s) { float e = expf(s_part[s] - mx); s_al[s] = e; den += e; }
        float inv = 1.0f / den;
        for (int s = 0; s < SS; ++s) s_al[s] *= inv;
    }
    __syncthreads();

    for (int j = tid; j < MM; j += 512) {   // ctx = al @ memory
        const float* mr = memory + (size_t)b * (SS * MM) + j;
        float c = 0.0f;
        for (int s = 0; s < SS; ++s) c = fmaf(s_al[s], mr[s * MM], c);
        s_ctx[j] = c;
    }
    __syncthreads();

    float a = 0.0f;                          // attn2 = [h2|ctx] @ Wa
    for (int k = 0; k < HH; ++k) a = fmaf(s_h2[k],  Wa[k * HH + tid], a);
    for (int k = 0; k < MM; ++k) a = fmaf(s_ctx[k], Wa[(HH + k) * HH + tid], a);

    attn_f32[t * (BB * HH) + b * HH + tid] = a;
    size_t r = (size_t)(t * BB + b) * HH + tid;     // packed A row (t*64+b)
    split_store(a, attn_hi, attn_lo, r);
}

// ---------------------------------------------------------------------------
// GEMM 3: logits = attn_all @ Wfc + bfc  [1984,512] x [512,34004]
// 8 waves/block, M-tile 128 (wave w -> m0 = by*128 + 16w), N-tile 64.
// Wfc packs are L2-resident; output is a 270 MB write-once stream ->
// non-temporal stores so it does not evict the packs from L2.
// ---------------------------------------------------------------------------
__global__ __launch_bounds__(256)
void logits_gemm(const __bf16* __restrict__ Ah, const __bf16* __restrict__ Al,
                 const __bf16* __restrict__ Bh, const __bf16* __restrict__ Bl,
                 const float* __restrict__ bfc, float* __restrict__ out) {
    int lane = threadIdx.x & 31, w = threadIdx.x >> 5;
    int m0 = blockIdx.y * 128 + w * 16;
    int n0 = blockIdx.x * 64;
    v8f acc[4] = {};
    for (int k0 = 0; k0 < HH; k0 += 32) {
        v16bf ah = frag_a(Ah, HH, m0, k0, lane);
        v16bf al = frag_a(Al, HH, m0, k0, lane);
#pragma unroll
        for (int j = 0; j < 4; ++j) {
            v16bf bh = frag_b(Bh, HH, n0 + j * 16, k0, lane);
            v16bf bl = frag_b(Bl, HH, n0 + j * 16, k0, lane);
            acc[j] = wmma3(acc[j], ah, al, bh, bl);
        }
    }
    int n = lane & 15, mb = (lane >> 4) << 3;
#pragma unroll
    for (int j = 0; j < 4; ++j)
#pragma unroll
        for (int v = 0; v < 8; ++v) {
            int r  = m0 + mb + v;
            int nn = n0 + j * 16 + n;
            if (r < RTOT && nn < VV) {
                int tt = r >> 6, bb = r & 63;           // r = t*64 + b
                __builtin_nontemporal_store(
                    acc[j][v] + bfc[nn],
                    &out[(size_t)bb * (TT * VV) + (size_t)tt * VV + nn]);
            }
        }
}

// ---------------------------------------------------------------------------
// Host-side launcher (graph-capture safe)
// ---------------------------------------------------------------------------
extern "C" void kernel_launch(void* const* d_in, const int* in_sizes, int n_in,
                              void* d_out, int out_size, void* d_ws, size_t ws_size,
                              hipStream_t stream) {
    const int*   dec    = (const int*)  d_in[0];
    const float* memory = (const float*)d_in[1];
    const float* h0     = (const float*)d_in[2];
    const float* c0     = (const float*)d_in[3];
    const float* emb    = (const float*)d_in[4];
    const float* Wk     = (const float*)d_in[5];
    const float* Wr     = (const float*)d_in[6];
    const float* bias   = (const float*)d_in[7];
    const float* Wm     = (const float*)d_in[8];
    const float* Wq     = (const float*)d_in[9];
    const float* vvec   = (const float*)d_in[10];
    const float* Wa     = (const float*)d_in[11];
    const float* Wfc    = (const float*)d_in[12];
    const float* bfc    = (const float*)d_in[13];

    // byte allocator over d_ws, 256B aligned (vector loads need 32B)
    char* wp = (char*)d_ws;
    auto alloc = [&](size_t bytes) -> char* {
        char* r = wp; wp += (bytes + 255) & ~(size_t)255; return r;
    };
    float*  keys     = (float*) alloc((size_t)BB * SS * HH * 4);   // 4 MB
    float*  zbuf     = (float*) alloc((size_t)BB * NZ * 4);        // 512 KB
    float*  h_state  = (float*) alloc((size_t)BB * HH * 4);
    float*  c_state  = (float*) alloc((size_t)BB * HH * 4);
    float*  attn_f32 = (float*) alloc((size_t)TT * BB * HH * 4);   // 4 MB
    __bf16* attn_hi  = (__bf16*)alloc((size_t)RPAD * HH * 2);      // 2 MB
    __bf16* attn_lo  = (__bf16*)alloc((size_t)RPAD * HH * 2);
    __bf16* memA_hi  = (__bf16*)alloc((size_t)BB * SS * MM * 2);   // 4 MB
    __bf16* memA_lo  = (__bf16*)alloc((size_t)BB * SS * MM * 2);
    __bf16* uhi      = (__bf16*)alloc((size_t)BB * KZP * 2);
    __bf16* ulo      = (__bf16*)alloc((size_t)BB * KZP * 2);
    __bf16* WmT_hi   = (__bf16*)alloc((size_t)HH * MM * 2);        // 1 MB
    __bf16* WmT_lo   = (__bf16*)alloc((size_t)HH * MM * 2);
    __bf16* WzT_hi   = (__bf16*)alloc((size_t)NZ * KZP * 2);       // 5.25 MB
    __bf16* WzT_lo   = (__bf16*)alloc((size_t)NZ * KZP * 2);
    __bf16* WfcT_hi  = (__bf16*)alloc((size_t)VVP * HH * 2);       // 33.25 MB
    __bf16* WfcT_lo  = (__bf16*)alloc((size_t)VVP * HH * 2);

    hipMemcpyAsync(h_state, h0, sizeof(float) * BB * HH, hipMemcpyDeviceToDevice, stream);
    hipMemcpyAsync(c_state, c0, sizeof(float) * BB * HH, hipMemcpyDeviceToDevice, stream);

    // ---- one-time packs -------------------------------------------------
    pack_rows<<<(BB * SS * MM + 255) / 256, 256, 0, stream>>>(memory, memA_hi, memA_lo, BB * SS * MM);
    pack_transpose<<<dim3(HH / 32, MM / 32), dim3(32, 8), 0, stream>>>(
        Wm, Wm, MM, MM, HH, MM, WmT_hi, WmT_lo);
    pack_transpose<<<dim3(NZ / 32, KZP / 32), dim3(32, 8), 0, stream>>>(
        Wk, Wr, CIN, KZ, NZ, KZP, WzT_hi, WzT_lo);
    pack_transpose<<<dim3(VVP / 32, HH / 32), dim3(32, 8), 0, stream>>>(
        Wfc, Wfc, HH, HH, VV, HH, WfcT_hi, WfcT_lo);

    // ---- keys = memory @ Wm (WMMA) --------------------------------------
    keys_gemm<<<dim3(HH / 64, (BB * SS) / 64), 128, 0, stream>>>(
        memA_hi, memA_lo, WmT_hi, WmT_lo, keys);

    // ---- sequential recurrence (logits deferred) ------------------------
    for (int t = 0; t < TT; ++t) {
        pack_u<<<(BB * KZP + 255) / 256, 256, 0, stream>>>(dec, emb, attn_f32, h_state, uhi, ulo, t);
        z_gemm<<<NZ / 64, 128, 0, stream>>>(uhi, ulo, WzT_hi, WzT_lo, zbuf);
        lstm_kernel<<<BB, HH, 0, stream>>>(zbuf, bias, h_state, c_state);
        attn_kernel<<<BB, HH, 0, stream>>>(h_state, keys, Wq, vvec, memory, Wa,
                                           attn_f32, attn_hi, attn_lo, t);
    }

    // ---- one big output projection (WMMA, Wfc pack L2-resident) ---------
    logits_gemm<<<dim3(VVP / 64, RPAD / 128), 256, 0, stream>>>(
        attn_hi, attn_lo, WfcT_hi, WfcT_lo, bfc, (float*)d_out);
}
